// SimpleKoopmanNet_7095285973362
// MI455X (gfx1250) — matrix operand order repaired and checked by reading, not compiled, not run on hardware
//
#include <hip/hip_runtime.h>

// ---------------------------------------------------------------------------
// SimpleKoopmanNet forward for MI455X (gfx1250, wave32, WMMA).
//   encoder MLP (bf16 WMMA GEMM + LN + tanh)  -> z0
//   batched expm(K*t) via f32 WMMA (scaling & squaring) -> zt
//   decoder MLP -> x_rec
// Outputs: d_out = [x_rec (B*8 f32) | zt (B*32 f32)]
// Workspace need: ~294 MB.
// ---------------------------------------------------------------------------

#define B_TOT 16384

typedef __attribute__((ext_vector_type(16))) __bf16        bf16x16;
typedef __attribute__((ext_vector_type(8)))  float         f32x8;
typedef __attribute__((ext_vector_type(2)))  float         f32x2;
typedef __attribute__((ext_vector_type(4)))  unsigned int  u32x4;

union FragAB { bf16x16 v; u32x4 q[2]; };

__device__ __forceinline__ unsigned short f2bf(float f) {
  unsigned int u = __float_as_uint(f);
  u = u + 0x7FFFu + ((u >> 16) & 1u);   // round-to-nearest-even
  return (unsigned short)(u >> 16);
}

// ---------------------------------------------------------------------------
// Weight prep: W f32 [Kin,Nin] (row-major) -> Wt bf16 [Npad,Kpad] (transposed,
// zero padded; Npad is a multiple of 64 so the GEMM needs no bounds checks).
// ---------------------------------------------------------------------------
__global__ __launch_bounds__(256)
void prep_w_kernel(const float* __restrict__ W, unsigned short* __restrict__ Wt,
                   int Kin, int Kpad, int Nin, int Npad) {
  long idx = (long)blockIdx.x * 256 + threadIdx.x;
  long tot = (long)Npad * Kpad;
  if (idx >= tot) return;
  int n = (int)(idx / Kpad);
  int k = (int)(idx - (long)n * Kpad);
  float v = (k < Kin && n < Nin) ? W[(size_t)k * Nin + n] : 0.0f;
  Wt[idx] = f2bf(v);
}

// x0 f32 [B,8] -> bf16 [B,32] zero-padded (K padded to one WMMA step)
__global__ __launch_bounds__(256)
void prep_x0_kernel(const float* __restrict__ x0, unsigned short* __restrict__ act) {
  long idx = (long)blockIdx.x * 256 + threadIdx.x;
  if (idx >= (long)B_TOT * 32) return;
  int b = (int)(idx >> 5), k = (int)(idx & 31);
  act[idx] = (k < 8) ? f2bf(x0[(size_t)b * 8 + k]) : (unsigned short)0;
}

// ---------------------------------------------------------------------------
// GEMM: C[B_TOT, ldc](f32) = A[B_TOT, K](bf16) x Wt[Npad, K](bf16)^T + bias
// 256 threads = 8 waves. Workgroup tile 256(M) x 64(N); wave tile 32x64
// (2 M-tiles x 4 N-tiles of v_wmma_f32_16x16x32_bf16).
// Software-pipelined K loop with two STATICALLY-named fragment buffers
// (dynamic indexing would force the buffers into scratch).
// Fragment layouts per CDNA5 ISA 7.12.2:
//   A 16x32 bf16: lane<16 holds K {k0..k0+7, k0+16..k0+23} of row M=lane;
//                 lane>=16 holds K {+8..+15, +24..+31}  -> two b128 loads.
//   B 32x16 bf16: lane<16 holds K {k0..k0+7, k0+16..k0+23} of col N=lane&15;
//                 lane>=16 holds K {+8..+15, +24..+31}  -> two b128 loads.
// ---------------------------------------------------------------------------
struct Frags { FragAB a0, a1, b[4]; };

__device__ __forceinline__ void load_frags(Frags& f,
                                           const unsigned short* __restrict__ aPtr0,
                                           const unsigned short* __restrict__ aPtr1,
                                           const unsigned short* const bPtr[4],
                                           int k0) {
  f.a0.q[0] = *(const u32x4*)(aPtr0 + k0);
  f.a0.q[1] = *(const u32x4*)(aPtr0 + k0 + 16);
  f.a1.q[0] = *(const u32x4*)(aPtr1 + k0);
  f.a1.q[1] = *(const u32x4*)(aPtr1 + k0 + 16);
#pragma unroll
  for (int nt = 0; nt < 4; ++nt) {
    f.b[nt].q[0] = *(const u32x4*)(bPtr[nt] + k0);
    f.b[nt].q[1] = *(const u32x4*)(bPtr[nt] + k0 + 16);
  }
}

__device__ __forceinline__ void mma_frags(f32x8 acc[2][4], const Frags& f) {
#pragma unroll
  for (int nt = 0; nt < 4; ++nt) {
    acc[0][nt] = __builtin_amdgcn_wmma_f32_16x16x32_bf16(
        false, f.a0.v, false, f.b[nt].v, (short)0, acc[0][nt], false, false);
    acc[1][nt] = __builtin_amdgcn_wmma_f32_16x16x32_bf16(
        false, f.a1.v, false, f.b[nt].v, (short)0, acc[1][nt], false, false);
  }
}

__global__ __launch_bounds__(256)
void koop_gemm_kernel(const unsigned short* __restrict__ A,
                      const unsigned short* __restrict__ Wt,
                      const float* __restrict__ bias,
                      float* __restrict__ C,
                      int K, int Nstore, int ldc) {
  const int lane = threadIdx.x & 31;
  const int wave = threadIdx.x >> 5;
  const int l16  = lane & 15;
  const int hi   = lane >> 4;

  const size_t rowBase = (size_t)blockIdx.y * 256 + (size_t)wave * 32;
  const int    colBase = blockIdx.x * 64;

  f32x8 acc[2][4] = {};

  const unsigned short* aPtr0 = A + (rowBase + l16) * (size_t)K + hi * 8;
  const unsigned short* aPtr1 = aPtr0 + (size_t)16 * K;

  const unsigned short* bPtr[4];
#pragma unroll
  for (int nt = 0; nt < 4; ++nt) {
    int col = colBase + nt * 16 + l16;          // always < Npad (padded alloc)
    bPtr[nt] = Wt + (size_t)col * K + hi * 8;
  }

  // Software pipeline, statically unrolled x2 so both buffers stay in VGPRs.
  Frags fr0, fr1;
  load_frags(fr0, aPtr0, aPtr1, bPtr, 0);
  int k0 = 32;
  for (; k0 + 32 < K; k0 += 64) {
    load_frags(fr1, aPtr0, aPtr1, bPtr, k0);
    mma_frags(acc, fr0);
    load_frags(fr0, aPtr0, aPtr1, bPtr, k0 + 32);
    mma_frags(acc, fr1);
  }
  if (k0 < K) {                 // even tile count: one tile still unloaded
    load_frags(fr1, aPtr0, aPtr1, bPtr, k0);
    mma_frags(acc, fr0);
    mma_frags(acc, fr1);
  } else {                      // odd tile count
    mma_frags(acc, fr0);
  }

  // Epilogue: D layout -> row = tileRow + hi*8 + r, col = tileCol + l16
#pragma unroll
  for (int nt = 0; nt < 4; ++nt) {
    int col = colBase + nt * 16 + l16;
    if (col >= Nstore) continue;
    float bv = bias[col];
#pragma unroll
    for (int mt = 0; mt < 2; ++mt) {
      size_t r0 = rowBase + (size_t)mt * 16 + (size_t)hi * 8;
      float* cp = C + r0 * (size_t)ldc + col;
#pragma unroll
      for (int r = 0; r < 8; ++r)
        cp[(size_t)r * ldc] = acc[mt][nt][r] + bv;
    }
  }
}

// ---------------------------------------------------------------------------
// LayerNorm + tanh: Y f32 [B,N] -> bf16 [B,N]. One row per workgroup.
// ---------------------------------------------------------------------------
__global__ __launch_bounds__(256)
void ln_tanh_kernel(const float* __restrict__ Y, const float* __restrict__ g,
                    const float* __restrict__ be, unsigned short* __restrict__ out,
                    int N) {
  __shared__ float s1[256];
  __shared__ float s2[256];
  const size_t row = blockIdx.x;
  const float* y = Y + row * (size_t)N;
  float a = 0.0f, b = 0.0f;
  for (int c = threadIdx.x; c < N; c += 256) {
    float v = y[c];
    a += v; b += v * v;
  }
  s1[threadIdx.x] = a; s2[threadIdx.x] = b;
  __syncthreads();
  for (int off = 128; off > 0; off >>= 1) {
    if ((int)threadIdx.x < off) {
      s1[threadIdx.x] += s1[threadIdx.x + off];
      s2[threadIdx.x] += s2[threadIdx.x + off];
    }
    __syncthreads();
  }
  float mean = s1[0] / (float)N;
  float var  = s2[0] / (float)N - mean * mean;
  float rinv = rsqrtf(var + 1e-5f);
  unsigned short* o = out + row * (size_t)N;
  for (int c = threadIdx.x; c < N; c += 256) {
    float v = (y[c] - mean) * rinv * g[c] + be[c];
    o[c] = f2bf(tanhf(v));
  }
}

// ---------------------------------------------------------------------------
// Batched expm of 32x32 matrices via f32 WMMA (16x16x4) on per-wave LDS tiles.
// One wave per matrix; 4 waves per block (48 KB static LDS).
// expm(Kt): scale by 2^-8, order-8 Taylor (Horner), square 8 times.
// Then zt = expm(Kt) @ z0;  writes zt f32 (d_out) + bf16 (decoder input).
// ---------------------------------------------------------------------------
#define EXPM_S     8
#define EXPM_ORDER 8

__device__ __forceinline__ void wave_mm32(const float* __restrict__ A,
                                          const float* __restrict__ Bm,
                                          float* __restrict__ D, int lane) {
  const int l16 = lane & 15;
  const int hi  = lane >> 4;
#pragma unroll
  for (int mi = 0; mi < 2; ++mi) {
#pragma unroll
    for (int ni = 0; ni < 2; ++ni) {
      f32x8 acc = {};
#pragma unroll
      for (int k0 = 0; k0 < 32; k0 += 4) {
        // A 16x4 f32: lane<16 -> K {k0,k0+1}; lane>=16 -> K {k0+2,k0+3}
        int ks = k0 + hi * 2;
        f32x2 a, b;
        a.x = A[(mi * 16 + l16) * 32 + ks];
        a.y = A[(mi * 16 + l16) * 32 + ks + 1];
        // B 4x16 f32: column = ni*16 + l16, same K split
        b.x = Bm[ks * 32 + ni * 16 + l16];
        b.y = Bm[(ks + 1) * 32 + ni * 16 + l16];
        acc = __builtin_amdgcn_wmma_f32_16x16x4_f32(
            false, a, false, b, (short)0, acc, false, false);
      }
#pragma unroll
      for (int r = 0; r < 8; ++r)
        D[(mi * 16 + hi * 8 + r) * 32 + ni * 16 + l16] = acc[r];
    }
  }
}

__global__ __launch_bounds__(128)
void expm_kernel(const float* __restrict__ Kd, const float* __restrict__ Koff,
                 const float* __restrict__ t, const float* __restrict__ z0,
                 float* __restrict__ zt_f32, unsigned short* __restrict__ zt_bf16) {
  __shared__ float sX[4][1024];
  __shared__ float sR[4][1024];
  __shared__ float sT[4][1024];
  const int lane = threadIdx.x & 31;
  const int wv   = threadIdx.x >> 5;
  const int mat  = blockIdx.x * 4 + wv;

  float* X = sX[wv];
  float* R = sR[wv];
  float* T = sT[wv];

  const float sc = t[mat] * (1.0f / (float)(1 << EXPM_S));
  // X = (K*eye + Koff*(1-eye)) * t / 2^s ; R = I + X/order
  for (int i = lane; i < 1024; i += 32) {
    int r = i >> 5, c = i & 31;
    float kv = (r == c) ? Kd[i] : Koff[i];
    float x = kv * sc;
    X[i] = x;
    R[i] = x * (1.0f / (float)EXPM_ORDER) + ((r == c) ? 1.0f : 0.0f);
  }
  // Horner: R = I + (X*R)/j   (all LDS ops are in-order within a wave)
  for (int j = EXPM_ORDER - 1; j >= 1; --j) {
    wave_mm32(X, R, T, lane);
    float invj = 1.0f / (float)j;
    for (int i = lane; i < 1024; i += 32) {
      int r = i >> 5, c = i & 31;
      R[i] = T[i] * invj + ((r == c) ? 1.0f : 0.0f);
    }
  }
  // Squarings
  for (int s = 0; s < EXPM_S; ++s) {
    wave_mm32(R, R, T, lane);
    for (int i = lane; i < 1024; i += 32) R[i] = T[i];
  }
  // zt[lane] = sum_c R[lane][c] * z0[c]
  const float* zr = z0 + (size_t)mat * 32;
  float acc = 0.0f;
#pragma unroll
  for (int c = 0; c < 32; ++c) acc += R[lane * 32 + c] * zr[c];
  zt_f32[(size_t)mat * 32 + lane]  = acc;
  zt_bf16[(size_t)mat * 32 + lane] = f2bf(acc);
}

// ---------------------------------------------------------------------------
// Host orchestration
// ---------------------------------------------------------------------------
extern "C" void kernel_launch(void* const* d_in, const int* in_sizes, int n_in,
                              void* d_out, int out_size, void* d_ws, size_t ws_size,
                              hipStream_t stream) {
  (void)in_sizes; (void)n_in; (void)out_size; (void)ws_size;
  char* ws = (char*)d_ws;

  // 12 GEMM weights: {W input index, Kin, Kpad, Nin, Npad}; Npad % 64 == 0.
  struct WPrep { int wIn, Kin, Kpad, Nin, Npad; };
  static const WPrep wp[12] = {
      {2,  8,    32,   1024, 1024},  // enc L0
      {6,  1024, 1024, 2048, 2048},  // enc L1
      {10, 2048, 2048, 2048, 2048},  // enc L2
      {14, 2048, 2048, 1024, 1024},  // enc L3
      {18, 1024, 1024, 512,  512 },  // enc L4
      {22, 512,  512,  32,   64  },  // enc out (N padded 32->64)
      {26, 32,   32,   512,  512 },  // dec L0
      {30, 512,  512,  1024, 1024},  // dec L1
      {34, 1024, 1024, 2048, 2048},  // dec L2
      {38, 2048, 2048, 2048, 2048},  // dec L3
      {42, 2048, 2048, 1024, 1024},  // dec L4
      {46, 1024, 1024, 8,    64  },  // dec out (N padded 8->64)
  };

  auto alignUp = [](size_t v) { return (v + 255) & ~(size_t)255; };
  size_t wtOff[12], off = 0;
  for (int i = 0; i < 12; ++i) {
    wtOff[i] = off;
    off = alignUp(off + (size_t)wp[i].Npad * wp[i].Kpad * 2);
  }
  size_t z0Off = off;  off = alignUp(off + (size_t)B_TOT * 32 * 4);
  size_t aOff  = off;  off = alignUp(off + (size_t)B_TOT * 2048 * 2);
  size_t bOff  = off;  off = alignUp(off + (size_t)B_TOT * 2048 * 2);
  size_t yOff  = off;  // + B*2048*4  => total ~294 MB

  unsigned short* wt[12];
  for (int i = 0; i < 12; ++i) wt[i] = (unsigned short*)(ws + wtOff[i]);
  float*          z0f  = (float*)(ws + z0Off);
  unsigned short* actA = (unsigned short*)(ws + aOff);
  unsigned short* actB = (unsigned short*)(ws + bOff);
  float*          Y    = (float*)(ws + yOff);
  float*          xrec = (float*)d_out;
  float*          ztO  = (float*)d_out + (size_t)B_TOT * 8;

  // --- weight prep (f32 -> transposed padded bf16) ---
  for (int i = 0; i < 12; ++i) {
    long n = (long)wp[i].Npad * wp[i].Kpad;
    prep_w_kernel<<<(unsigned)((n + 255) / 256), 256, 0, stream>>>(
        (const float*)d_in[wp[i].wIn], wt[i],
        wp[i].Kin, wp[i].Kpad, wp[i].Nin, wp[i].Npad);
  }
  prep_x0_kernel<<<(B_TOT * 32 + 255) / 256, 256, 0, stream>>>(
      (const float*)d_in[0], actA);

  auto GEMM = [&](const unsigned short* Aact, unsigned short* w, int bIdx,
                  float* Cout, int K, int Npad, int Nstore, int ldc) {
    dim3 grid((unsigned)(Npad / 64), B_TOT / 256);
    koop_gemm_kernel<<<grid, 256, 0, stream>>>(
        Aact, w, (const float*)d_in[bIdx], Cout, K, Nstore, ldc);
  };
  auto LN = [&](const float* Yb, int gIdx, int beIdx, unsigned short* outb, int N) {
    ln_tanh_kernel<<<B_TOT, 256, 0, stream>>>(
        Yb, (const float*)d_in[gIdx], (const float*)d_in[beIdx], outb, N);
  };

  // --- encoder ---
  GEMM(actA, wt[0], 3,  Y,   32,   1024, 1024, 1024); LN(Y, 4,  5,  actB, 1024);
  GEMM(actB, wt[1], 7,  Y,   1024, 2048, 2048, 2048); LN(Y, 8,  9,  actA, 2048);
  GEMM(actA, wt[2], 11, Y,   2048, 2048, 2048, 2048); LN(Y, 12, 13, actB, 2048);
  GEMM(actB, wt[3], 15, Y,   2048, 1024, 1024, 1024); LN(Y, 16, 17, actA, 1024);
  GEMM(actA, wt[4], 19, Y,   1024, 512,  512,  512 ); LN(Y, 20, 21, actB, 512);
  GEMM(actB, wt[5], 23, z0f, 512,  64,   32,   32  );

  // --- Koopman propagation: zt = expm(K_full * t) @ z0 ---
  expm_kernel<<<B_TOT / 4, 128, 0, stream>>>(
      (const float*)d_in[24], (const float*)d_in[25], (const float*)d_in[1],
      z0f, ztO, actA);

  // --- decoder ---
  GEMM(actA, wt[6],  27, Y,    32,   512,  512,  512 ); LN(Y, 28, 29, actB, 512);
  GEMM(actB, wt[7],  31, Y,    512,  1024, 1024, 1024); LN(Y, 32, 33, actA, 1024);
  GEMM(actA, wt[8],  35, Y,    1024, 2048, 2048, 2048); LN(Y, 36, 37, actB, 2048);
  GEMM(actB, wt[9],  39, Y,    2048, 2048, 2048, 2048); LN(Y, 40, 41, actA, 2048);
  GEMM(actA, wt[10], 43, Y,    2048, 1024, 1024, 1024); LN(Y, 44, 45, actB, 1024);
  GEMM(actB, wt[11], 47, xrec, 1024, 64,   8,    8   );
}
